// MultiHeadedAttention_38998303047888
// MI455X (gfx1250) — matrix operand order repaired and checked by reading, not compiled, not run on hardware
//
#include <hip/hip_runtime.h>

#define B_ 2
#define S_ 2048
#define D_ 1024
#define H_ 16
#define DK_ 64

typedef __attribute__((ext_vector_type(16))) __bf16 v16bf;
typedef __attribute__((ext_vector_type(2)))  __bf16 bf16x2;
typedef __attribute__((ext_vector_type(8)))  float  v8f;
typedef __attribute__((ext_vector_type(4)))  unsigned v4u;
typedef __attribute__((ext_vector_type(8)))  int    v8i;
typedef __attribute__((ext_vector_type(4)))  int    v4i;

union FragBF { v16bf v; unsigned u[8]; };

__device__ __forceinline__ unsigned short f2bf(float x) {
  __bf16 b = (__bf16)x;                      // native v_cvt, RNE
  return __builtin_bit_cast(unsigned short, b);
}
__device__ __forceinline__ unsigned pk_bf16(float a, float b) {
  bf16x2 t; t[0] = (__bf16)a; t[1] = (__bf16)b;
  return __builtin_bit_cast(unsigned, t);
}
// A-fragment (16-bit, 16x32) per-lane K offset for register r
__device__ __forceinline__ int kA0(int r, int hlf) {
  return ((r >> 2) << 4) + (hlf << 3) + ((r & 3) << 1);
}
// B-fragment (16-bit, 32x16) per-lane K offset for register r
__device__ __forceinline__ int kB0(int r, int hlf) {
  return ((r >> 2) << 4) + (hlf << 3) + ((r & 3) << 1);
}

// ---- TDM: 2D tile (16-bit elements) global -> LDS.  D# per ISA ch.8. ----
__device__ __forceinline__ void tdm_load_2d(unsigned lds_off, const void* gptr,
                                            unsigned tensor_d0, unsigned tensor_d1,
                                            unsigned tile_d0, unsigned tile_d1,
                                            unsigned long long stride0_elems) {
  unsigned long long ga = (unsigned long long)gptr;
  v4u g0;
  g0[0] = 1u;                                           // count=1, user-mode, no gather
  g0[1] = lds_off;                                      // D#.lds_addr (bytes)
  g0[2] = (unsigned)ga;                                 // global_addr low 32
  g0[3] = (unsigned)((ga >> 32) & 0x1FFFFFFu) | (2u << 30);  // addr hi | type=2
  v8i g1;
  g1[0] = (int)(1u << 16);                              // data_size=1 (2 bytes), mask=0
  g1[1] = (int)((tensor_d0 & 0xFFFFu) << 16);           // tensor_dim0[15:0]
  g1[2] = (int)((tensor_d0 >> 16) | ((tensor_d1 & 0xFFFFu) << 16));
  g1[3] = (int)((tensor_d1 >> 16) | (tile_d0 << 16));   // tile_dim0
  g1[4] = (int)(tile_d1);                               // tile_dim1 (tile_dim2=0)
  g1[5] = (int)(unsigned)(stride0_elems & 0xFFFFFFFFull);
  g1[6] = (int)(unsigned)((stride0_elems >> 32) & 0xFFFFull);   // stride1=0
  g1[7] = 0;
  v4i z4 = {0, 0, 0, 0};
  v8i z8 = {0, 0, 0, 0, 0, 0, 0, 0};
  __builtin_amdgcn_tensor_load_to_lds(g0, g1, z4, z4, z8, 0);   // 6-arg toolchain form
}

// ---- Weight prep: Wq/Wk/Wv [H,D,DK] f32 -> [H,DK,D] bf16 (scale folded) ----
__global__ void k_wt_proj(const float* __restrict__ W, unsigned short* __restrict__ WT, float scale) {
  int idx = blockIdx.x * blockDim.x + threadIdx.x;   // over H*DK*D, layout [h][k][d]
  int d = idx & (D_ - 1);
  int t = idx >> 10;
  int k = t & (DK_ - 1);
  int h = t >> 6;
  WT[idx] = f2bf(W[((size_t)h * D_ + d) * DK_ + k] * scale);
}

// ---- Wu [D,D] f32 -> WuT [n][c] bf16 ----
__global__ void k_wt_u(const float* __restrict__ W, unsigned short* __restrict__ WT) {
  int idx = blockIdx.x * blockDim.x + threadIdx.x;   // over D*D, layout [n][c]
  int c = idx & (D_ - 1);
  int n = idx >> 10;
  WT[idx] = f2bf(W[(size_t)c * D_ + n]);
}

// ---- Projection GEMM: Out[b,h,s,k] (or vT: Out[b,h,k,s]) = In[b,s,:] x W[h,:,:] ----
__global__ void __launch_bounds__(128)
k_proj(const float* __restrict__ In, const unsigned short* __restrict__ WT,
       unsigned short* __restrict__ Out, int transposed) {
  const int lane = threadIdx.x & 31;
  const int wave = threadIdx.x >> 5;
  const int hlf  = lane >> 4;
  const int mn   = lane & 15;
  const int b = blockIdx.z, h = blockIdx.y;
  const int s0 = blockIdx.x * 64 + wave * 16;

  const float* arow = In + ((size_t)b * S_ + s0 + mn) * D_;
  const unsigned short* wh = WT + (size_t)h * DK_ * D_;

  v8f acc[4] = {};
  for (int d0 = 0; d0 < D_; d0 += 32) {
    FragBF a;
#pragma unroll
    for (int r = 0; r < 8; ++r) {
      float2 qv = *(const float2*)(arow + d0 + kA0(r, hlf));
      a.u[r] = pk_bf16(qv.x, qv.y);
    }
#pragma unroll
    for (int nt = 0; nt < 4; ++nt) {
      const unsigned short* brow = wh + (size_t)(nt * 16 + mn) * D_ + d0;
      FragBF bf;
#pragma unroll
      for (int r = 0; r < 8; ++r) bf.u[r] = *(const unsigned*)(brow + kB0(r, hlf));
      acc[nt] = __builtin_amdgcn_wmma_f32_16x16x32_bf16(false, a.v, false, bf.v,
                                                        (short)0, acc[nt], false, false);
    }
  }
  if (!transposed) {           // Out [b,h,s,dk]
#pragma unroll
    for (int nt = 0; nt < 4; ++nt) {
      int kd = nt * 16 + mn;
#pragma unroll
      for (int r = 0; r < 8; ++r) {
        int s = s0 + (hlf << 3) + r;
        Out[(((size_t)b * H_ + h) * S_ + s) * DK_ + kd] = f2bf(acc[nt][r]);
      }
    }
  } else {                     // vT: Out [b,h,k,s]
#pragma unroll
    for (int nt = 0; nt < 4; ++nt) {
      int kd = nt * 16 + mn;
      unsigned* orow = (unsigned*)(Out + (((size_t)b * H_ + h) * DK_ + kd) * (size_t)S_
                                   + s0 + (hlf << 3));
#pragma unroll
      for (int r = 0; r < 8; r += 2)
        orow[r >> 1] = pk_bf16(acc[nt][r], acc[nt][r + 1]);
    }
  }
}

// ---- Flash attention: TDM double-buffers K/VT tiles into LDS; WMMA compute ----
__global__ void __launch_bounds__(128)
k_attn(const unsigned short* __restrict__ Q, const unsigned short* __restrict__ K,
       const unsigned short* __restrict__ VT, const int* __restrict__ Msk,
       unsigned short* __restrict__ Oa) {
  __shared__ __align__(16) unsigned short kTile[2][32][64];   // 8 KB
  __shared__ __align__(16) unsigned short vTile[2][64][32];   // 8 KB
  __shared__ __align__(16) unsigned short lds_ps[4][16][32];  // 4 KB p-transpose
  const int lane = threadIdx.x & 31;
  const int wave = threadIdx.x >> 5;
  const int hlf  = lane >> 4;
  const int mn   = lane & 15;
  const int b = blockIdx.z, h = blockIdx.y;
  const int s0 = blockIdx.x * 64 + wave * 16;

  const unsigned short* qbase  = Q  + (((size_t)b * H_ + h) * S_) * DK_;
  const unsigned short* kbase  = K  + (((size_t)b * H_ + h) * S_) * DK_;
  const unsigned short* vtbase = VT + (((size_t)b * H_ + h) * DK_) * S_;
  const int* mbase = Msk + (size_t)b * S_ * S_;

  FragBF qf[2];
  {
    const unsigned short* qrow = qbase + (size_t)(s0 + mn) * DK_;
#pragma unroll
    for (int kk = 0; kk < 2; ++kk)
#pragma unroll
      for (int r = 0; r < 8; ++r)
        qf[kk].u[r] = *(const unsigned*)(qrow + kk * 32 + kA0(r, hlf));
  }

  // kick off DMA for first tile pair (wave 0 only; TENSORcnt tracks it)
  if (wave == 0) {
    // K rows t0..t0+31 are contiguous (32*64 bf16): 1-D copy
    tdm_load_2d((unsigned)(size_t)&kTile[0][0][0], kbase, 2048u, 1u, 2048u, 1u, 2048ull);
    // VT tile: 64 rows x 32 cols, row stride S_
    tdm_load_2d((unsigned)(size_t)&vTile[0][0][0], vtbase, 32u, 64u, 32u, 64u, (unsigned long long)S_);
  }

  v8f acc[4] = {};
  float mrun[8], lrun[8];
#pragma unroll
  for (int r = 0; r < 8; ++r) { mrun[r] = -3.0e38f; lrun[r] = 0.f; }

  int buf = 0;
  for (int t0 = 0; t0 < S_; t0 += 32, buf ^= 1) {
    if (wave == 0) __builtin_amdgcn_s_wait_tensorcnt(0);   // current buffers landed
    __syncthreads();                                        // publish; prior step done
    if (wave == 0 && t0 + 32 < S_) {                        // prefetch next tiles via TDM
      tdm_load_2d((unsigned)(size_t)&kTile[buf ^ 1][0][0],
                  kbase + (size_t)(t0 + 32) * DK_, 2048u, 1u, 2048u, 1u, 2048ull);
      tdm_load_2d((unsigned)(size_t)&vTile[buf ^ 1][0][0],
                  vtbase + (t0 + 32), 32u, 64u, 32u, 64u, (unsigned long long)S_);
    }

    // -------- scores: two 16x16 tiles (t0, t0+16), K over DK=64 --------
    v8f sf[2];
#pragma unroll
    for (int tt = 0; tt < 2; ++tt) {
      const unsigned short* krow = &kTile[buf][tt * 16 + mn][0];
      FragBF kf0, kf1;
#pragma unroll
      for (int r = 0; r < 8; ++r) {
        int k0 = kB0(r, hlf);
        kf0.u[r] = *(const unsigned*)(krow + k0);
        kf1.u[r] = *(const unsigned*)(krow + 32 + k0);
      }
      v8f c = {};
      c = __builtin_amdgcn_wmma_f32_16x16x32_bf16(false, qf[0].v, false, kf0.v, (short)0, c, false, false);
      c = __builtin_amdgcn_wmma_f32_16x16x32_bf16(false, qf[1].v, false, kf1.v, (short)0, c, false, false);
      sf[tt] = c;
    }
    // -------- mask (32 MB int mask lives in the 192 MB L2) --------
#pragma unroll
    for (int tt = 0; tt < 2; ++tt) {
      int tcol = t0 + tt * 16 + mn;
#pragma unroll
      for (int r = 0; r < 8; ++r) {
        int srow = s0 + (hlf << 3) + r;
        if (mbase[(size_t)srow * S_ + tcol] == 0) sf[tt][r] = -1.0e9f;
      }
    }
    // -------- online softmax (row reductions across 16-lane groups) --------
    float fac[8];
#pragma unroll
    for (int r = 0; r < 8; ++r) {
      float mx = fmaxf(sf[0][r], sf[1][r]);
      mx = fmaxf(mx, __shfl_xor(mx, 1, 32));
      mx = fmaxf(mx, __shfl_xor(mx, 2, 32));
      mx = fmaxf(mx, __shfl_xor(mx, 4, 32));
      mx = fmaxf(mx, __shfl_xor(mx, 8, 32));
      float mnew = fmaxf(mrun[r], mx);
      fac[r] = __expf(mrun[r] - mnew);
      mrun[r] = mnew;
      float p0 = __expf(sf[0][r] - mnew);
      float p1 = __expf(sf[1][r] - mnew);
      sf[0][r] = p0; sf[1][r] = p1;
      float rs = p0 + p1;
      rs += __shfl_xor(rs, 1, 32);
      rs += __shfl_xor(rs, 2, 32);
      rs += __shfl_xor(rs, 4, 32);
      rs += __shfl_xor(rs, 8, 32);
      lrun[r] = lrun[r] * fac[r] + rs;
    }
    // -------- transpose p (C layout -> A layout) through LDS --------
#pragma unroll
    for (int tt = 0; tt < 2; ++tt)
#pragma unroll
      for (int r = 0; r < 8; ++r)
        lds_ps[wave][(hlf << 3) + r][tt * 16 + mn] = f2bf(sf[tt][r]);

    FragBF pf;
#pragma unroll
    for (int r = 0; r < 8; ++r)
      pf.u[r] = *(const unsigned*)&lds_ps[wave][mn][kA0(r, hlf)];

    // -------- rescale accumulators, then acc += p x V --------
#pragma unroll
    for (int nt = 0; nt < 4; ++nt)
#pragma unroll
      for (int r = 0; r < 8; ++r) acc[nt][r] *= fac[r];
#pragma unroll
    for (int nt = 0; nt < 4; ++nt) {
      const unsigned short* vrow = &vTile[buf][nt * 16 + mn][0];
      FragBF vf;
#pragma unroll
      for (int r = 0; r < 8; ++r) vf.u[r] = *(const unsigned*)(vrow + kB0(r, hlf));
      acc[nt] = __builtin_amdgcn_wmma_f32_16x16x32_bf16(false, pf.v, false, vf.v,
                                                        (short)0, acc[nt], false, false);
    }
  }
  // -------- normalize and store heads concatenated: Oa[b,s, h*DK+kd] --------
#pragma unroll
  for (int nt = 0; nt < 4; ++nt) {
    int kd = nt * 16 + mn;
#pragma unroll
    for (int r = 0; r < 8; ++r) {
      int srow = s0 + (hlf << 3) + r;
      Oa[((size_t)b * S_ + srow) * D_ + h * DK_ + kd] = f2bf(acc[nt][r] / lrun[r]);
    }
  }
}

// ---- Unify: Out[r, n] = X[r,:] x WuT[n,:] + bu[n]  (f32 out) ----
__global__ void __launch_bounds__(128)
k_unify(const unsigned short* __restrict__ X, const unsigned short* __restrict__ WuT,
        const float* __restrict__ bu, float* __restrict__ Out) {
  const int lane = threadIdx.x & 31;
  const int wave = threadIdx.x >> 5;
  const int hlf  = lane >> 4;
  const int mn   = lane & 15;
  const int row0 = blockIdx.x * 64 + wave * 16;   // over B*S
  const int n0   = blockIdx.y * 64;

  const unsigned short* xrow = X + (size_t)(row0 + mn) * D_;
  v8f acc[4] = {};
  for (int d0 = 0; d0 < D_; d0 += 32) {
    FragBF a;
#pragma unroll
    for (int r = 0; r < 8; ++r) a.u[r] = *(const unsigned*)(xrow + d0 + kA0(r, hlf));
#pragma unroll
    for (int nt = 0; nt < 4; ++nt) {
      const unsigned short* wrow = WuT + (size_t)(n0 + nt * 16 + mn) * D_ + d0;
      FragBF bf;
#pragma unroll
      for (int r = 0; r < 8; ++r) bf.u[r] = *(const unsigned*)(wrow + kB0(r, hlf));
      acc[nt] = __builtin_amdgcn_wmma_f32_16x16x32_bf16(false, a.v, false, bf.v,
                                                        (short)0, acc[nt], false, false);
    }
  }
#pragma unroll
  for (int nt = 0; nt < 4; ++nt) {
    int col = n0 + nt * 16 + mn;
    float bias = bu[col];
#pragma unroll
    for (int r = 0; r < 8; ++r) {
      int row = row0 + (hlf << 3) + r;
      Out[(size_t)row * D_ + col] = acc[nt][r] + bias;
    }
  }
}

extern "C" void kernel_launch(void* const* d_in, const int* in_sizes, int n_in,
                              void* d_out, int out_size, void* d_ws, size_t ws_size,
                              hipStream_t stream) {
  (void)in_sizes; (void)n_in; (void)out_size; (void)ws_size;
  const float* query = (const float*)d_in[0];
  const float* key_  = (const float*)d_in[1];
  const float* value = (const float*)d_in[2];
  const int*   mask  = (const int*)d_in[3];
  const float* Wq = (const float*)d_in[4];
  const float* Wk = (const float*)d_in[5];
  const float* Wv = (const float*)d_in[6];
  const float* Wu = (const float*)d_in[7];
  const float* bu = (const float*)d_in[8];
  float* out = (float*)d_out;

  char* ws = (char*)d_ws;
  size_t off = 0;
  auto alloc = [&](size_t bytes) -> char* {
    char* p = ws + off; off += (bytes + 255) & ~(size_t)255; return p;
  };
  unsigned short* WqT = (unsigned short*)alloc((size_t)H_ * DK_ * D_ * 2);
  unsigned short* WkT = (unsigned short*)alloc((size_t)H_ * DK_ * D_ * 2);
  unsigned short* WvT = (unsigned short*)alloc((size_t)H_ * DK_ * D_ * 2);
  unsigned short* WuT = (unsigned short*)alloc((size_t)D_ * D_ * 2);
  unsigned short* Qb  = (unsigned short*)alloc((size_t)B_ * H_ * S_ * DK_ * 2);
  unsigned short* Kb  = (unsigned short*)alloc((size_t)B_ * H_ * S_ * DK_ * 2);
  unsigned short* VTb = (unsigned short*)alloc((size_t)B_ * H_ * DK_ * S_ * 2);
  unsigned short* Xb  = (unsigned short*)alloc((size_t)B_ * S_ * D_ * 2);

  const float qkscale = 0.35355339059327373f;   // 64^(-1/4)
  int wtn = (H_ * DK_ * D_) / 256;
  k_wt_proj<<<wtn, 256, 0, stream>>>(Wq, WqT, qkscale);
  k_wt_proj<<<wtn, 256, 0, stream>>>(Wk, WkT, qkscale);
  k_wt_proj<<<wtn, 256, 0, stream>>>(Wv, WvT, 1.0f);
  k_wt_u<<<(D_ * D_) / 256, 256, 0, stream>>>(Wu, WuT);

  dim3 gp(S_ / 64, H_, B_);
  k_proj<<<gp, 128, 0, stream>>>(query, WqT, Qb, 0);
  k_proj<<<gp, 128, 0, stream>>>(key_,  WkT, Kb, 0);
  k_proj<<<gp, 128, 0, stream>>>(value, WvT, VTb, 1);

  dim3 ga(S_ / 64, H_, B_);
  k_attn<<<ga, 128, 0, stream>>>(Qb, Kb, VTb, mask, Xb);

  dim3 gu((B_ * S_) / 64, D_ / 64, 1);
  k_unify<<<gu, 128, 0, stream>>>(Xb, WuT, bu, out);
}
